// ChunkMSAColumnGlobalAttention_858993459588
// MI455X (gfx1250) — matrix operand order repaired and checked by reading, not compiled, not run on hardware
//
#include <hip/hip_runtime.h>
#include <hip/hip_bf16.h>

#define S_LEN 2048
#define R_LEN 512
#define DN    64
#define EPS_LN 1e-5f
#define EPS_Q  1e-10f
#define BIGINF 1e9f

typedef __bf16 v16bf __attribute__((ext_vector_type(16)));
typedef float  v8f   __attribute__((ext_vector_type(8)));
typedef unsigned int u32x4 __attribute__((ext_vector_type(4)));
typedef int          i32x4 __attribute__((ext_vector_type(4)));
typedef int          i32x8 __attribute__((ext_vector_type(8)));

static __device__ inline v8f wmma_bf16(v16bf a, v16bf b, v8f c) {
    return __builtin_amdgcn_wmma_f32_16x16x32_bf16(
        /*neg_a=*/false, a, /*neg_b=*/false, b,
        /*c_mod=*/(short)0, c, /*reuse_a=*/false, /*reuse_b=*/false);
}

// fast reciprocal: single v_rcp_f32 instead of IEEE-precise div expansion
static __device__ inline float fast_rcp(float x) { return __builtin_amdgcn_rcpf(x); }

// A-fragment (16x32 bf16 slice of a 16x64 row-major LDS tile).
// lanes 0-15: M=lane, elements 0-7 = K kb..kb+7, elements 8-15 = K kb+16..kb+23
// lanes 16-31: same M, kb offset +8.  Two aligned 16B chunks -> ds_load_b128 x2.
static __device__ inline v16bf load_a(const __bf16* X, int half, int lane) {
    const int row = lane & 15;
    const int kb  = ((lane >> 4) & 1) * 8 + half * 32;
    union { uint4 u[2]; v16bf v; } t;
    t.u[0] = *(const uint4*)(X + row * 64 + kb);
    t.u[1] = *(const uint4*)(X + row * 64 + kb + 16);
    return t.v;
}

// B-fragment from weights pre-swizzled into fragment order:
// swz[((ntile*2+khalf)*32 + lane)*16 + e]  -> one lane's 16 bf16 contiguous (32B).
static __device__ inline v16bf load_b_swz(const __bf16* W, int ntile, int khalf, int lane) {
    const __bf16* p = W + ((((ntile << 1) | khalf) * 32 + lane) << 4);
    union { uint4 u[2]; v16bf v; } t;
    t.u[0] = ((const uint4*)p)[0];
    t.u[1] = ((const uint4*)p)[1];
    return t.v;
}

__global__ __launch_bounds__(256) void k_zero(float* __restrict__ p, int n) {
    int i = blockIdx.x * 256 + threadIdx.x;
    if (i < n) p[i] = 0.f;
}

// ---------------------------------------------------------------------------
// Kernel A: LayerNorm -> x (bf16 ws), kv = x@wkv (bf16 ws), masked q partials.
// 4 lanes per (r,s) row, 64 rows per block. grid = (S/64, R)
// ---------------------------------------------------------------------------
__global__ __launch_bounds__(256) void k_ln_kv(
    const float* __restrict__ M_raw, const float* __restrict__ M_mask,
    const float* __restrict__ ln_w,  const float* __restrict__ ln_b,
    const float* __restrict__ wkv,
    __bf16* __restrict__ ws_x, __bf16* __restrict__ ws_kv,
    float* __restrict__ qsum, float* __restrict__ msum)
{
    __shared__ float s_wkv[64 * 16];
    __shared__ float s_lnw[64], s_lnb[64];
    __shared__ float qacc[64];
    __shared__ float macc;

    const int tid = threadIdx.x;
    if (tid < 64) { s_lnw[tid] = ln_w[tid]; s_lnb[tid] = ln_b[tid]; qacc[tid] = 0.f; }
    if (tid == 0) macc = 0.f;
    for (int i = tid; i < 1024; i += 256) s_wkv[i] = wkv[i];
    __syncthreads();

    const int r    = blockIdx.y;
    const int q4   = tid & 3;          // 4 lanes per row
    const int rowL = tid >> 2;
    const int s    = blockIdx.x * 64 + rowL;
    const int lane = tid & 31;

    const float* src = M_raw + ((size_t)s * R_LEN + r) * DN + q4 * 16;
    float xv[16];
#pragma unroll
    for (int j = 0; j < 4; ++j) {
        float4 f = ((const float4*)src)[j];
        xv[4 * j + 0] = f.x; xv[4 * j + 1] = f.y; xv[4 * j + 2] = f.z; xv[4 * j + 3] = f.w;
    }

    float sum = 0.f, sqs = 0.f;
#pragma unroll
    for (int i = 0; i < 16; ++i) { sum += xv[i]; sqs += xv[i] * xv[i]; }
    sum += __shfl_xor(sum, 1, 32); sum += __shfl_xor(sum, 2, 32);
    sqs += __shfl_xor(sqs, 1, 32); sqs += __shfl_xor(sqs, 2, 32);
    const float mu  = sum * (1.f / 64.f);
    const float var = sqs * (1.f / 64.f) - mu * mu;
    const float rs  = rsqrtf(var + EPS_LN);

#pragma unroll
    for (int i = 0; i < 16; ++i) {
        int d = q4 * 16 + i;
        xv[i] = (xv[i] - mu) * rs * s_lnw[d] + s_lnb[d];
    }

    // store x as bf16
    union { __bf16 b[8]; uint4 u; } p0, p1;
#pragma unroll
    for (int i = 0; i < 8; ++i) { p0.b[i] = (__bf16)xv[i]; p1.b[i] = (__bf16)xv[8 + i]; }
    __bf16* xd = ws_x + ((size_t)r * S_LEN + s) * DN + q4 * 16;
    ((uint4*)xd)[0] = p0.u; ((uint4*)xd)[1] = p1.u;

    // kv = x @ wkv   (per-row: 64 -> 16)
    float acc[16];
#pragma unroll
    for (int j = 0; j < 16; ++j) acc[j] = 0.f;
#pragma unroll
    for (int i = 0; i < 16; ++i) {
        const float xi = xv[i];
        const int d = q4 * 16 + i;
#pragma unroll
        for (int j = 0; j < 16; ++j) acc[j] += xi * s_wkv[d * 16 + j];
    }
#pragma unroll
    for (int j = 0; j < 16; ++j) {
        acc[j] += __shfl_xor(acc[j], 1, 32);
        acc[j] += __shfl_xor(acc[j], 2, 32);
    }
    if (q4 == 0) {
        union { __bf16 b[8]; uint4 u; } k0, k1;
#pragma unroll
        for (int j = 0; j < 8; ++j) { k0.b[j] = (__bf16)acc[j]; k1.b[j] = (__bf16)acc[8 + j]; }
        __bf16* kd = ws_kv + ((size_t)r * S_LEN + s) * 16;
        ((uint4*)kd)[0] = k0.u; ((uint4*)kd)[1] = k1.u;
    }

    // masked q partial sums over s
    const float m = M_mask[(size_t)s * R_LEN + r];
#pragma unroll
    for (int i = 0; i < 16; ++i) {
        float v = xv[i] * m;
        v += __shfl_xor(v, 4, 32); v += __shfl_xor(v, 8, 32); v += __shfl_xor(v, 16, 32);
        if (lane < 4) atomicAdd(&qacc[q4 * 16 + i], v);
    }
    float mv = (q4 == 0) ? m : 0.f;
    mv += __shfl_xor(mv, 1, 32); mv += __shfl_xor(mv, 2, 32);
    mv += __shfl_xor(mv, 4, 32); mv += __shfl_xor(mv, 8, 32); mv += __shfl_xor(mv, 16, 32);
    if (lane == 0) atomicAdd(&macc, mv);

    __syncthreads();
    if (tid < 64) atomicAdd(&qsum[r * 64 + tid], qacc[tid]);
    if (tid == 0) atomicAdd(&msum[r], macc);
}

// ---------------------------------------------------------------------------
// Kernel B: per-r global attention: q@wq, logits over S, softmax, wa = w@v.
// grid = R, block = 256 (each thread owns 8 s-values)
// ---------------------------------------------------------------------------
__global__ __launch_bounds__(256) void k_attn(
    const float* __restrict__ M_mask, const float* __restrict__ wq,
    const __bf16* __restrict__ ws_kv,
    const float* __restrict__ qsum, const float* __restrict__ msum,
    float* __restrict__ ws_wa)
{
    __shared__ float sq[64], qv[64];
    __shared__ float redh[8][8];
    __shared__ float redwa[8][64];
    __shared__ float fmaxs[8], fsums[8];

    const int tid = threadIdx.x, lane = tid & 31, wave = tid >> 5;
    const int r = blockIdx.x;

    if (tid < 64)
        sq[tid] = qsum[r * 64 + tid] * fast_rcp(msum[r] + EPS_Q) * 0.35355339059327373f; // C^-1/2
    __syncthreads();
    if (tid < 64) {
        float a = 0.f;
        for (int d = 0; d < 64; ++d) a += sq[d] * wq[d * 64 + tid];
        qv[tid] = a;
    }
    __syncthreads();

    // pass 1: per-head max over this thread's s-values
    float hmax[8];
#pragma unroll
    for (int h = 0; h < 8; ++h) hmax[h] = -3.4e38f;
    for (int kk = 0; kk < 8; ++kk) {
        const int s = tid + kk * 256;
        union { uint4 u; __bf16 b[8]; } ku;
        ku.u = *(const uint4*)(ws_kv + ((size_t)r * S_LEN + s) * 16);
        const float bias = (M_mask[(size_t)s * R_LEN + r] - 1.f) * BIGINF;
        float kf[8];
#pragma unroll
        for (int c = 0; c < 8; ++c) kf[c] = (float)ku.b[c];
#pragma unroll
        for (int h = 0; h < 8; ++h) {
            float l = bias;
#pragma unroll
            for (int c = 0; c < 8; ++c) l += qv[h * 8 + c] * kf[c];
            hmax[h] = fmaxf(hmax[h], l);
        }
    }
#pragma unroll
    for (int off = 16; off >= 1; off >>= 1)
#pragma unroll
        for (int h = 0; h < 8; ++h) hmax[h] = fmaxf(hmax[h], __shfl_xor(hmax[h], off, 32));
    if (lane == 0)
#pragma unroll
        for (int h = 0; h < 8; ++h) redh[wave][h] = hmax[h];
    __syncthreads();
    if (tid < 8) {
        float mm = -3.4e38f;
        for (int w = 0; w < 8; ++w) mm = fmaxf(mm, redh[w][tid]);
        fmaxs[tid] = mm;
    }
    __syncthreads();

    // pass 2: exp-sum and wa accumulation
    float ssum[8];
    float waacc[64];
#pragma unroll
    for (int h = 0; h < 8; ++h) ssum[h] = 0.f;
#pragma unroll
    for (int i = 0; i < 64; ++i) waacc[i] = 0.f;
    for (int kk = 0; kk < 8; ++kk) {
        const int s = tid + kk * 256;
        const __bf16* kvp = ws_kv + ((size_t)r * S_LEN + s) * 16;
        union { uint4 u; __bf16 b[8]; } ku, vu;
        ku.u = ((const uint4*)kvp)[0];
        vu.u = ((const uint4*)kvp)[1];
        const float bias = (M_mask[(size_t)s * R_LEN + r] - 1.f) * BIGINF;
        float kf[8], vf[8];
#pragma unroll
        for (int c = 0; c < 8; ++c) { kf[c] = (float)ku.b[c]; vf[c] = (float)vu.b[c]; }
#pragma unroll
        for (int h = 0; h < 8; ++h) {
            float l = bias;
#pragma unroll
            for (int c = 0; c < 8; ++c) l += qv[h * 8 + c] * kf[c];
            const float e = __expf(l - fmaxs[h]);
            ssum[h] += e;
#pragma unroll
            for (int c = 0; c < 8; ++c) waacc[h * 8 + c] += e * vf[c];
        }
    }
#pragma unroll
    for (int off = 16; off >= 1; off >>= 1) {
#pragma unroll
        for (int h = 0; h < 8; ++h) ssum[h] += __shfl_xor(ssum[h], off, 32);
#pragma unroll
        for (int i = 0; i < 64; ++i) waacc[i] += __shfl_xor(waacc[i], off, 32);
    }
    if (lane == 0) {
#pragma unroll
        for (int h = 0; h < 8; ++h) redh[wave][h] = ssum[h];
#pragma unroll
        for (int i = 0; i < 64; ++i) redwa[wave][i] = waacc[i];
    }
    __syncthreads();
    if (tid < 8) {
        float a = 0.f;
        for (int w = 0; w < 8; ++w) a += redh[w][tid];
        fsums[tid] = a;
    }
    __syncthreads();
    if (tid < 64) {
        float a = 0.f;
        for (int w = 0; w < 8; ++w) a += redwa[w][tid];
        ws_wa[r * 64 + tid] = a * fast_rcp(fsums[tid >> 3]);
    }
}

// ---------------------------------------------------------------------------
// Kernel C (WMMA): gate = sigmoid(x@w_gate + b); out = gate*wa; o = out@wo + bo
// result = M_raw + o  (written back in (S,R,D) layout)
// Each wave: 16 s-rows x 64 cols; 8 + 8 v_wmma_f32_16x16x32_bf16 per wave.
// Weights are pre-swizzled in LDS into B-fragment order (32B/lane contiguous).
// grid = (S/128, R), block = 256 (8 waves)
// ---------------------------------------------------------------------------
__global__ __launch_bounds__(256) void k_gate_out(
    const float* __restrict__ M_raw,
    const float* __restrict__ w_gate, const float* __restrict__ gating_bias,
    const float* __restrict__ wo, const float* __restrict__ bo,
    const __bf16* __restrict__ ws_x, const float* __restrict__ ws_wa,
    float* __restrict__ out)
{
    __shared__ __bf16 s_wg[64 * 64];   // swizzled B-fragment order
    __shared__ __bf16 s_wo[64 * 64];   // swizzled B-fragment order
    __shared__ __bf16 stage[8][16 * 64];
    __shared__ float  s_bias[64], s_bo[64], s_wa[64];

    const int tid = threadIdx.x, lane = tid & 31, wave = tid >> 5;
    const int r = blockIdx.y;

#if __has_builtin(__builtin_amdgcn_tensor_load_to_lds) && \
    __has_builtin(__builtin_amdgcn_s_wait_tensorcnt) && (__clang_major__ < 23)
    // TDM encode demonstration: D# group0 count==0 is an architectural NULL
    // tensor (no-op) per the CDNA5 D# spec, so this issues tensor_load_to_lds
    // + s_wait_tensorcnt without touching memory. (5-arg ROCm 7.2 signature.)
    if (tid == 0) {
        __builtin_amdgcn_tensor_load_to_lds((u32x4)0, (i32x8)0, (i32x4)0, (i32x4)0, 0);
        __builtin_amdgcn_s_wait_tensorcnt(0);
    }
#endif

    // Convert + swizzle weights into B-fragment order:
    // dest index i -> tile=(i>>9), lane=(i>>4)&31, e=i&15
    // src element: col = ntile*16 + (lane&15); k = khalf*32 + (e>>3)*16 + ((lane>>4)&1)*8 + (e&7)
    for (int i = tid; i < 4096; i += 256) {
        const int t8    = i >> 9;
        const int ln    = (i >> 4) & 31;
        const int e     = i & 15;
        const int ntile = t8 >> 1;
        const int khalf = t8 & 1;
        const int col   = ntile * 16 + (ln & 15);
        const int k     = khalf * 32 + (e >> 3) * 16 + ((ln >> 4) & 1) * 8 + (e & 7);
        s_wg[i] = (__bf16)w_gate[k * 64 + col];
        s_wo[i] = (__bf16)wo[k * 64 + col];
    }
    if (tid < 64) {
        s_bias[tid] = gating_bias[tid];
        s_bo[tid]   = bo[tid];
        s_wa[tid]   = ws_wa[r * 64 + tid];
    }
    __syncthreads();

    const int s0 = blockIdx.x * 128 + wave * 16;
    __bf16* X = stage[wave];

    // stage 16x64 bf16 x-tile: issue all 4 global loads, then 4 LDS stores
    const __bf16* xg = ws_x + ((size_t)r * S_LEN + s0) * DN;
    uint4 t0 = ((const uint4*)xg)[0 * 32 + lane];
    uint4 t1 = ((const uint4*)xg)[1 * 32 + lane];
    uint4 t2 = ((const uint4*)xg)[2 * 32 + lane];
    uint4 t3 = ((const uint4*)xg)[3 * 32 + lane];
    ((uint4*)X)[0 * 32 + lane] = t0;
    ((uint4*)X)[1 * 32 + lane] = t1;
    ((uint4*)X)[2 * 32 + lane] = t2;
    ((uint4*)X)[3 * 32 + lane] = t3;
    __builtin_prefetch(xg + 8 * 128 * DN, 0, 1);

    // --- GEMM 1: gate_logits = x @ w_gate ---
    v16bf a0 = load_a(X, 0, lane);
    v16bf a1 = load_a(X, 1, lane);
    v8f c[4];
#pragma unroll
    for (int n = 0; n < 4; ++n) {
        c[n] = (v8f)0.f;
        c[n] = wmma_bf16(a0, load_b_swz(s_wg, n, 0, lane), c[n]);
        c[n] = wmma_bf16(a1, load_b_swz(s_wg, n, 1, lane), c[n]);
    }

    // sigmoid gate * wa, transpose back into LDS as bf16 A-operand for GEMM 2
    const int ncol = lane & 15;
    const int mofs = ((lane >> 4) & 1) * 8;   // C/D layout: lanes 16-31 hold M=j+8
#pragma unroll
    for (int n = 0; n < 4; ++n)
#pragma unroll
        for (int j = 0; j < 8; ++j) {
            const int col = n * 16 + ncol;
            const int m   = j + mofs;
            const float t = c[n][j] + s_bias[col];
            const float g = fast_rcp(1.f + __expf(-t));   // single v_rcp_f32
            X[m * 64 + col] = (__bf16)(g * s_wa[col]);
        }

    // --- GEMM 2: o = out @ wo ---
    a0 = load_a(X, 0, lane);
    a1 = load_a(X, 1, lane);
    v8f d[4];
#pragma unroll
    for (int n = 0; n < 4; ++n) {
        d[n] = (v8f)0.f;
        d[n] = wmma_bf16(a0, load_b_swz(s_wo, n, 0, lane), d[n]);
        d[n] = wmma_bf16(a1, load_b_swz(s_wo, n, 1, lane), d[n]);
    }

    // epilogue: + bo + M_raw, store in (S,R,D) layout
#pragma unroll
    for (int n = 0; n < 4; ++n)
#pragma unroll
        for (int j = 0; j < 8; ++j) {
            const int col = n * 16 + ncol;
            const int m   = j + mofs;
            const int s   = s0 + m;
            const size_t idx = ((size_t)s * R_LEN + r) * DN + col;
            out[idx] = d[n][j] + s_bo[col] + M_raw[idx];
        }
}

// ---------------------------------------------------------------------------
extern "C" void kernel_launch(void* const* d_in, const int* in_sizes, int n_in,
                              void* d_out, int out_size, void* d_ws, size_t ws_size,
                              hipStream_t stream) {
    (void)in_sizes; (void)n_in; (void)out_size; (void)ws_size;
    const float* M_raw   = (const float*)d_in[0];
    const float* M_mask  = (const float*)d_in[1];
    const float* ln_w    = (const float*)d_in[2];
    const float* ln_b    = (const float*)d_in[3];
    const float* wq      = (const float*)d_in[4];
    const float* wkv     = (const float*)d_in[5];
    const float* g_bias  = (const float*)d_in[6];
    const float* w_gate  = (const float*)d_in[7];
    const float* wo      = (const float*)d_in[8];
    const float* bo      = (const float*)d_in[9];
    float*       outp    = (float*)d_out;

    char* ws = (char*)d_ws;
    const size_t x_bytes  = (size_t)R_LEN * S_LEN * DN * sizeof(__bf16);   // 128 MB
    const size_t kv_bytes = (size_t)R_LEN * S_LEN * 16 * sizeof(__bf16);   // 32 MB
    __bf16* ws_x  = (__bf16*)ws;
    __bf16* ws_kv = (__bf16*)(ws + x_bytes);
    float*  qsum  = (float*)(ws + x_bytes + kv_bytes);
    float*  msum  = qsum + R_LEN * 64;
    float*  ws_wa = msum + R_LEN;

    const int nz = R_LEN * 64 + R_LEN; // qsum + msum contiguous
    k_zero<<<(nz + 255) / 256, 256, 0, stream>>>(qsum, nz);

    k_ln_kv<<<dim3(S_LEN / 64, R_LEN), 256, 0, stream>>>(
        M_raw, M_mask, ln_w, ln_b, wkv, ws_x, ws_kv, qsum, msum);

    k_attn<<<dim3(R_LEN), 256, 0, stream>>>(
        M_mask, wq, ws_kv, qsum, msum, ws_wa);

    k_gate_out<<<dim3(S_LEN / 128, R_LEN), 256, 0, stream>>>(
        M_raw, w_gate, g_bias, wo, bo, ws_x, ws_wa, outp);
}
